// MIM_65661460021328
// MI455X (gfx1250) — compile-verified, matrix operand order checked by reading, not run on hardware
//
#include <hip/hip_runtime.h>
#include <hip/hip_bf16.h>
#include <utility>

typedef _Float16 half_t;
typedef __attribute__((ext_vector_type(16))) _Float16 v16h;
typedef __attribute__((ext_vector_type(8)))  _Float16 v8h;
typedef __attribute__((ext_vector_type(8)))  float    v8f;

#define NHID   64
#define HWD    64
#define HW     4096          // 64*64
#define BATCH  4
#define SITES  16384         // BATCH*HW
#define TSTEPS 19
#define ILEN   10

// ---------------------------------------------------------------------------
// Weight packing: OIHW f32 -> [tap][Cout][Cin] f16
// ---------------------------------------------------------------------------
__global__ void pack_tap(const float* __restrict__ w, half_t* __restrict__ out,
                         int Cout, int Cin, int k2) {
  int total = Cout * Cin * k2;
  for (int idx = blockIdx.x * 256 + threadIdx.x; idx < total;
       idx += gridDim.x * 256) {
    int ci  = idx % Cin;
    int r   = idx / Cin;
    int co  = r % Cout;
    int tap = r / Cout;
    out[idx] = (half_t)w[((size_t)co * Cin + ci) * k2 + tap];
  }
}

__global__ void zero_u32(unsigned* __restrict__ p, size_t n) {
  size_t i  = (size_t)blockIdx.x * blockDim.x + threadIdx.x;
  size_t st = (size_t)gridDim.x * blockDim.x;
  for (; i < n; i += st) p[i] = 0u;
}

// ---------------------------------------------------------------------------
// WMMA implicit-GEMM conv: 5x5 SAME (TAPS=25) or 1x1 (TAPS=1).
// in : f16 NHWC [B,64,64,CIN]   (CIN multiple of 32)
// wp : f16 [TAPS][Cout][CIN]
// out: f32 NHWC [B,64,64,Cout]  (Cout multiple of 64), bias fused
// grid = (B*64, Cout/64), block = 64 (2 waves); each wave: 32(M) x 64(N),
// i.e. two A fragments amortized over the four B fragments of one image row.
// ---------------------------------------------------------------------------
template <int TAPS, int CIN>
__global__ __launch_bounds__(64) void conv_wmma(
    const half_t* __restrict__ in, const half_t* __restrict__ wp,
    const float* __restrict__ bias, float* __restrict__ out, int Cout) {
  const int lane = threadIdx.x & 31;
  const int wave = threadIdx.x >> 5;
  const int lr   = lane & 15;
  const int hi8  = (lane >> 4) << 3;        // 0 or 8 : K sub-offset / D row off
  const int m0   = (blockIdx.y * 2 + wave) * 32;
  const int row  = blockIdx.x;              // b*64 + y
  const int b    = row >> 6;
  const int y    = row & 63;

  v8f acc0[4] = {};
  v8f acc1[4] = {};
  const v8h zv = {};

#pragma unroll 1
  for (int tap = 0; tap < TAPS; ++tap) {
    const int dy = (TAPS == 25) ? (tap / 5 - 2) : 0;
    const int dx = (TAPS == 25) ? (tap % 5 - 2) : 0;
    const int yy = y + dy;
    if ((unsigned)yy >= 64u) continue;
    const size_t wbase0 = ((size_t)tap * Cout + (m0 + lr)) * CIN + hi8;
    const size_t wbase1 = wbase0 + (size_t)16 * CIN;
    const size_t sbase  = (size_t)((b << 6) + yy) << 6;  // site base (x added)
#pragma unroll
    for (int kc = 0; kc < CIN; kc += 32) {
      union { v16h v; v8h h[2]; } A0, A1;
      const v8h* w0 = (const v8h*)(wp + wbase0 + kc);
      const v8h* w1 = (const v8h*)(wp + wbase1 + kc);
      A0.h[0] = w0[0]; A0.h[1] = w0[2];      // K = kc+hi8(+16) .. +8
      A1.h[0] = w1[0]; A1.h[1] = w1[2];
#pragma unroll
      for (int t = 0; t < 4; ++t) {
        union { v16h v; v8h h[2]; } Bm;
        const int xx = t * 16 + lr + dx;
        if ((unsigned)xx < 64u) {
          const v8h* ipv = (const v8h*)(in + (sbase + xx) * CIN + kc + hi8);
          Bm.h[0] = ipv[0];
          Bm.h[1] = ipv[2];
        } else {
          Bm.h[0] = zv;
          Bm.h[1] = zv;
        }
        acc0[t] = __builtin_amdgcn_wmma_f32_16x16x32_f16(
            false, A0.v, false, Bm.v, (short)0, acc0[t], false, false);
        acc1[t] = __builtin_amdgcn_wmma_f32_16x16x32_f16(
            false, A1.v, false, Bm.v, (short)0, acc1[t], false, false);
      }
    }
  }
#pragma unroll
  for (int t = 0; t < 4; ++t) {
    const int xo = t * 16 + lr;
    const size_t so = ((size_t)row * 64 + xo) * Cout;
#pragma unroll
    for (int v = 0; v < 8; ++v) {
      const int co0 = m0 + v + hi8;
      const int co1 = co0 + 16;
      out[so + co0] = acc0[t][v] + bias[co0];
      out[so + co1] = acc1[t][v] + bias[co1];
    }
  }
}

// ---------------------------------------------------------------------------
// Direct conv for Cin=1, k=5, Cout=256 (stlstm x-branch; ~0.2 GFLOP, VALU ok)
// ---------------------------------------------------------------------------
__global__ __launch_bounds__(256) void conv_cin1(
    const half_t* __restrict__ x, const float* __restrict__ w,
    const float* __restrict__ bias, float* __restrict__ out) {
  const int total = SITES * 256;
  for (int idx = blockIdx.x * 256 + threadIdx.x; idx < total;
       idx += gridDim.x * 256) {
    const int co   = idx & 255;
    const int site = idx >> 8;
    const int b  = site >> 12;
    const int yy = (site >> 6) & 63;
    const int xx = site & 63;
    float s = bias[co];
    const float* wr = w + co * 25;
#pragma unroll
    for (int ky = 0; ky < 5; ++ky) {
      const int y2 = yy + ky - 2;
      if ((unsigned)y2 >= 64u) continue;
#pragma unroll
      for (int kx = 0; kx < 5; ++kx) {
        const int x2 = xx + kx - 2;
        if ((unsigned)x2 >= 64u) continue;
        s += wr[ky * 5 + kx] * (float)x[(b << 12) + (y2 << 6) + x2];
      }
    }
    out[(size_t)site * 256 + co] = s;
  }
}

// ---------------------------------------------------------------------------
// LayerNorm stats: deterministic two-stage reduction over each batch sample
// ---------------------------------------------------------------------------
__global__ __launch_bounds__(256) void ln_part(const float* __restrict__ buf,
                                               int perB,
                                               float* __restrict__ part) {
  __shared__ float s0[256], s1[256];
  const int b = blockIdx.y, p = blockIdx.x;
  const int chunk = perB >> 6;                 // NPART = 64
  const float* base = buf + (size_t)b * perB + (size_t)p * chunk;
  float a = 0.f, q = 0.f;
  for (int i = threadIdx.x; i < chunk; i += 256) {
    float v = base[i]; a += v; q += v * v;
  }
  s0[threadIdx.x] = a; s1[threadIdx.x] = q; __syncthreads();
  for (int o = 128; o; o >>= 1) {
    if (threadIdx.x < o) { s0[threadIdx.x] += s0[threadIdx.x + o];
                           s1[threadIdx.x] += s1[threadIdx.x + o]; }
    __syncthreads();
  }
  if (threadIdx.x == 0) {
    part[(b * 64 + p) * 2 + 0] = s0[0];
    part[(b * 64 + p) * 2 + 1] = s1[0];
  }
}

__global__ __launch_bounds__(64) void ln_final(const float* __restrict__ part,
                                               float* __restrict__ stats,
                                               int perB) {
  __shared__ float s0[64], s1[64];
  const int b = blockIdx.x, i = threadIdx.x;
  s0[i] = part[(b * 64 + i) * 2 + 0];
  s1[i] = part[(b * 64 + i) * 2 + 1];
  __syncthreads();
  for (int o = 32; o; o >>= 1) {
    if (i < o) { s0[i] += s0[i + o]; s1[i] += s1[i + o]; }
    __syncthreads();
  }
  if (i == 0) {
    float mu  = s0[0] / (float)perB;
    float var = s1[0] / (float)perB - mu * mu;
    stats[2 * b]     = mu;
    stats[2 * b + 1] = rsqrtf(var + 1e-6f);
  }
}

// ---------------------------------------------------------------------------
// Elementwise gate kernels (LN affine fused)
// ---------------------------------------------------------------------------
__device__ __forceinline__ float sigm(float x) {
  return 1.f / (1.f + __expf(-x));
}
__device__ __forceinline__ float lnv(const float* __restrict__ cc,
                                     const float* __restrict__ st,
                                     const float* __restrict__ g,
                                     const float* __restrict__ be,
                                     size_t sC, int b, int ch) {
  return (cc[sC + ch] - st[2 * b]) * st[2 * b + 1] * g[ch] + be[ch];
}

__global__ __launch_bounds__(256) void stlstm_gates(
    const float* __restrict__ xcc, const float* __restrict__ tcc,
    const float* __restrict__ scc, const float* __restrict__ stx,
    const float* __restrict__ stt, const float* __restrict__ sts,
    const float* __restrict__ gx_, const float* __restrict__ bx_,
    const float* __restrict__ gt_, const float* __restrict__ bt_,
    const float* __restrict__ gs_, const float* __restrict__ bs_,
    const half_t* __restrict__ cold, const half_t* __restrict__ mold,
    half_t* __restrict__ cnew, half_t* __restrict__ mnew,
    half_t* __restrict__ cat, half_t* __restrict__ obuf) {
  const int tid = blockIdx.x * 256 + threadIdx.x;
  const int site = tid >> 6, c = tid & 63, b = site >> 12;
  const size_t sC = (size_t)site * 256;
  float ix = lnv(xcc, stx, gx_, bx_, sC, b, c);
  float gx = lnv(xcc, stx, gx_, bx_, sC, b, 64 + c);
  float fx = lnv(xcc, stx, gx_, bx_, sC, b, 128 + c);
  float ox = lnv(xcc, stx, gx_, bx_, sC, b, 192 + c);
  float it = lnv(tcc, stt, gt_, bt_, sC, b, c);
  float gt = lnv(tcc, stt, gt_, bt_, sC, b, 64 + c);
  float ft = lnv(tcc, stt, gt_, bt_, sC, b, 128 + c);
  float ot = lnv(tcc, stt, gt_, bt_, sC, b, 192 + c);
  float is = lnv(scc, sts, gs_, bs_, sC, b, c);
  float gs = lnv(scc, sts, gs_, bs_, sC, b, 64 + c);
  float fs = lnv(scc, sts, gs_, bs_, sC, b, 128 + c);
  float os = lnv(scc, sts, gs_, bs_, sC, b, 192 + c);
  float i  = sigm(ix + it),  g  = tanhf(gx + gt);
  float i2 = sigm(ix + is),  g2 = tanhf(gx + gs);
  float f  = sigm(fx + ft + 1.f), f2 = sigm(fx + fs + 1.f);
  float o  = sigm(ox + ot + os);
  float nm = f2 * (float)mold[tid] + i2 * g2;
  float nc = f  * (float)cold[tid] + i  * g;
  cnew[tid] = (half_t)nc; mnew[tid] = (half_t)nm;
  const size_t cb = (size_t)site * 128;
  cat[cb + c] = (half_t)nc; cat[cb + 64 + c] = (half_t)nm;
  obuf[tid] = (half_t)o;
}

__global__ __launch_bounds__(256) void mimn_gates(
    const float* __restrict__ hcc, const float* __restrict__ xcc,
    const float* __restrict__ sth, const float* __restrict__ stx,
    const float* __restrict__ gh_, const float* __restrict__ bh_,
    const float* __restrict__ gx_, const float* __restrict__ bx_,
    const half_t* __restrict__ cold, const float* __restrict__ ctw,
    const float* __restrict__ ocw, half_t* __restrict__ cnew,
    half_t* __restrict__ hnew) {
  const int tid = blockIdx.x * 256 + threadIdx.x;
  const int site = tid >> 6, c = tid & 63, b = site >> 12, hw = site & 4095;
  const size_t sC = (size_t)site * 256;
  float ih = lnv(hcc, sth, gh_, bh_, sC, b, c);
  float gh = lnv(hcc, sth, gh_, bh_, sC, b, 64 + c);
  float fh = lnv(hcc, sth, gh_, bh_, sC, b, 128 + c);
  float oh = lnv(hcc, sth, gh_, bh_, sC, b, 192 + c);
  float ix = lnv(xcc, stx, gx_, bx_, sC, b, c);
  float gx = lnv(xcc, stx, gx_, bx_, sC, b, 64 + c);
  float fx = lnv(xcc, stx, gx_, bx_, sC, b, 128 + c);
  float ox = lnv(xcc, stx, gx_, bx_, sC, b, 192 + c);
  float co_ = (float)cold[tid];
  float ic = co_ * ctw[c * HW + hw];
  float fc = co_ * ctw[(64 + c) * HW + hw];
  float i2 = sigm(ih + ix + ic);
  float f2 = sigm(fh + fx + fc + 1.f);
  float g2 = tanhf(gh + gx);
  float cn = f2 * co_ + i2 * g2;
  float oc = cn * ocw[c * HW + hw];
  float hn = sigm(oh + ox + oc) * tanhf(cn);
  cnew[tid] = (half_t)cn;
  hnew[tid] = (half_t)hn;
}

__global__ __launch_bounds__(256) void mimblock_gates(
    const float* __restrict__ tcc, const float* __restrict__ scc,
    const float* __restrict__ xcc, const float* __restrict__ stt,
    const float* __restrict__ sts, const float* __restrict__ stx,
    const float* __restrict__ gt_, const float* __restrict__ bt_,
    const float* __restrict__ gs_, const float* __restrict__ bs_,
    const float* __restrict__ gx_, const float* __restrict__ bx_,
    const half_t* __restrict__ mold, half_t* __restrict__ mnew,
    half_t* __restrict__ cat, half_t* __restrict__ ig,
    half_t* __restrict__ obuf) {
  const int tid = blockIdx.x * 256 + threadIdx.x;
  const int site = tid >> 6, c = tid & 63, b = site >> 12;
  const size_t sC3 = (size_t)site * 192, sC = (size_t)site * 256;
  float it = lnv(tcc, stt, gt_, bt_, sC3, b, c);
  float gt = lnv(tcc, stt, gt_, bt_, sC3, b, 64 + c);
  float ot = lnv(tcc, stt, gt_, bt_, sC3, b, 128 + c);
  float is = lnv(scc, sts, gs_, bs_, sC, b, c);
  float gs = lnv(scc, sts, gs_, bs_, sC, b, 64 + c);
  float fs = lnv(scc, sts, gs_, bs_, sC, b, 128 + c);
  float os = lnv(scc, sts, gs_, bs_, sC, b, 192 + c);
  float ix = lnv(xcc, stx, gx_, bx_, sC, b, c);
  float gx = lnv(xcc, stx, gx_, bx_, sC, b, 64 + c);
  float fx = lnv(xcc, stx, gx_, bx_, sC, b, 128 + c);
  float ox = lnv(xcc, stx, gx_, bx_, sC, b, 192 + c);
  float i  = sigm(ix + it),  g  = tanhf(gx + gt);
  float i2 = sigm(ix + is),  g2 = tanhf(gx + gs);
  float f2 = sigm(fx + fs + 1.f);
  float o  = sigm(ox + ot + os);
  float nm = f2 * (float)mold[tid] + i2 * g2;
  mnew[tid] = (half_t)nm;
  cat[(size_t)site * 128 + 64 + c] = (half_t)nm;
  ig[tid]   = (half_t)(i * g);
  obuf[tid] = (half_t)o;
}

__global__ __launch_bounds__(256) void finish_mim(
    const half_t* __restrict__ cmims, const half_t* __restrict__ ig,
    half_t* __restrict__ cnew, half_t* __restrict__ cat) {
  const int tid = blockIdx.x * 256 + threadIdx.x;
  const int site = tid >> 6, c = tid & 63;
  float nc = (float)cmims[tid] + (float)ig[tid];
  cnew[tid] = (half_t)nc;
  cat[(size_t)site * 128 + c] = (half_t)nc;
}

__global__ __launch_bounds__(256) void apply_out(
    const half_t* __restrict__ obuf, const float* __restrict__ cell,
    half_t* __restrict__ hout) {
  const int tid = blockIdx.x * 256 + threadIdx.x;
  hout[tid] = (half_t)((float)obuf[tid] * tanhf(cell[tid]));
}

__global__ __launch_bounds__(256) void sub_f16(
    const half_t* __restrict__ a, const half_t* __restrict__ b,
    half_t* __restrict__ out) {
  const int tid = blockIdx.x * 256 + threadIdx.x;
  out[tid] = (half_t)((float)a[tid] - (float)b[tid]);
}

__global__ __launch_bounds__(256) void blend_x(
    const float* __restrict__ images, const float* __restrict__ ss,
    const half_t* __restrict__ xprev, half_t* __restrict__ xcur, int t) {
  const int i = blockIdx.x * 256 + threadIdx.x;     // SITES
  const int b = i >> 12, hw = i & 4095;
  float img = images[((size_t)(b * 20 + t)) * HW + hw];
  float a = 1.f;
  if (t >= ILEN) a = ss[((size_t)(b * (TSTEPS - ILEN)) + (t - ILEN)) * HW + hw];
  float xp = (float)xprev[i];
  xcur[i] = (half_t)(a * img + (1.f - a) * xp);
}

__global__ __launch_bounds__(256) void xgen_dot(
    const half_t* __restrict__ h, const float* __restrict__ w,
    const float* __restrict__ bias, float* __restrict__ out,
    half_t* __restrict__ xprev, int t) {
  const int i = blockIdx.x * 256 + threadIdx.x;     // SITES
  const int b = i >> 12, hw = i & 4095;
  const half_t* hp = h + (size_t)i * 64;
  float s = bias[0];
#pragma unroll
  for (int c = 0; c < 64; ++c) s += (float)hp[c] * w[c];
  out[((size_t)(b * TSTEPS + t)) * HW + hw] = s;
  xprev[i] = (half_t)s;
}

__global__ __launch_bounds__(256) void loss_part(
    const float* __restrict__ gen, const float* __restrict__ images,
    float* __restrict__ part) {
  __shared__ float sm[256];
  const int N = BATCH * TSTEPS * HW;
  float s = 0.f;
  for (int i = blockIdx.x * 256 + threadIdx.x; i < N; i += gridDim.x * 256) {
    int b = i / (TSTEPS * HW);
    int r = i - b * TSTEPS * HW;
    int t = r >> 12, hw = r & 4095;
    float d = gen[i] - images[((size_t)(b * 20 + t + 1)) * HW + hw];
    s += d * d;
  }
  sm[threadIdx.x] = s; __syncthreads();
  for (int o = 128; o; o >>= 1) {
    if (threadIdx.x < o) sm[threadIdx.x] += sm[threadIdx.x + o];
    __syncthreads();
  }
  if (threadIdx.x == 0) part[blockIdx.x] = sm[0];
}

__global__ __launch_bounds__(256) void loss_final(
    const float* __restrict__ part, float* __restrict__ out) {
  __shared__ float sm[256];
  sm[threadIdx.x] = part[threadIdx.x]; __syncthreads();
  for (int o = 128; o; o >>= 1) {
    if (threadIdx.x < o) sm[threadIdx.x] += sm[threadIdx.x + o];
    __syncthreads();
  }
  if (threadIdx.x == 0) out[0] = sm[0] / (float)(BATCH * TSTEPS * HW);
}

// ---------------------------------------------------------------------------
// Host orchestration
// ---------------------------------------------------------------------------
extern "C" void kernel_launch(void* const* d_in, const int* in_sizes, int n_in,
                              void* d_out, int out_size, void* d_ws,
                              size_t ws_size, hipStream_t stream) {
  (void)in_sizes; (void)ws_size;
  if (n_in < 120) return;
  const float* images = (const float*)d_in[0];
  const float* ss     = (const float*)d_in[1];
  auto F = [&](int i) { return (const float*)d_in[i]; };

  // pytree flatten order (dicts sorted by key): diff[3]x10, mim[3]x24,
  // stlstm x14, xgen x2
  const int DIFF = 2, MIM = 32, STL = 104, XGEN = 118;
  auto diffB = [&](int j) { return DIFF + 10 * j; };
  auto mimB  = [&](int j) { return MIM + 24 * j; };
  // mimn:  0 ct_w, 1 h.b, 2 h.w, 3 ln_h.b, 4 ln_h.g, 5 ln_x.b, 6 ln_x.g,
  //        7 oc_w, 8 x.b, 9 x.w
  // mim:   0 c.b, 1 c.w, 2 h.b, 3 h.w, 4 ln_h.b, 5 ln_h.g, 6 ln_m.b,
  //        7 ln_m.g, 8 ln_x.b, 9 ln_x.g, 10 m.b, 11 m.w, 12..21 mims,
  //        22 x.b, 23 x.w
  // stlstm:0 c.b, 1 c.w, 2 h.b, 3 h.w, 4 ln_h.b, 5 ln_h.g, 6 ln_m.b,
  //        7 ln_m.g, 8 ln_x.b, 9 ln_x.g, 10 m.b, 11 m.w, 12 x.b, 13 x.w

  // ---- workspace layout ----
  size_t cur = 0;
  auto alloc = [&](size_t bytes) -> char* {
    size_t a = (cur + 255) & ~(size_t)255;
    cur = a + bytes;
    return (char*)d_ws + a;
  };
  auto AH = [&](size_t n) { return (half_t*)alloc(n * sizeof(half_t)); };
  auto AF = [&](size_t n) { return (float*)alloc(n * sizeof(float)); };

  const size_t WC256 = (size_t)25 * 256 * 64;   // 5x5, Cout256, Cin64
  const size_t WC192 = (size_t)25 * 192 * 64;
  const size_t WC1x1 = (size_t)64 * 128;

  half_t* wp_stl_h = AH(WC256);
  half_t* wp_stl_m = AH(WC256);
  half_t* wp_stl_c = AH(WC1x1);
  half_t *wp_mim_x[3], *wp_mim_h[3], *wp_mim_m[3], *wp_mim_c[3];
  half_t *wp_mims_h[3], *wp_mims_x[3], *wp_diff_h[3], *wp_diff_x[3];
  for (int j = 0; j < 3; ++j) {
    wp_mim_x[j] = AH(WC256);  wp_mim_h[j] = AH(WC192);
    wp_mim_m[j] = AH(WC256);  wp_mim_c[j] = AH(WC1x1);
    wp_mims_h[j] = AH(WC256); wp_mims_x[j] = AH(WC256);
    wp_diff_h[j] = AH(WC256); wp_diff_x[j] = AH(WC256);
  }

  // zero-initialized state region
  const size_t SB = (size_t)SITES * NHID;       // f16 elems per state tensor
  size_t zstart = (cur + 255) & ~(size_t)255;
  half_t *hs_cur[4], *hs_nxt[4], *cs_cur[4], *cs_nxt[4];
  half_t *dhs_cur[3], *dhs_nxt[3], *dcs_cur[3], *dcs_nxt[3];
  half_t *mcs_cur[3], *mcs_nxt[3];
  for (int k = 0; k < 4; ++k) { hs_cur[k] = AH(SB); hs_nxt[k] = AH(SB);
                                cs_cur[k] = AH(SB); cs_nxt[k] = AH(SB); }
  for (int k = 0; k < 3; ++k) { dhs_cur[k] = AH(SB); dhs_nxt[k] = AH(SB);
                                dcs_cur[k] = AH(SB); dcs_nxt[k] = AH(SB);
                                mcs_cur[k] = AH(SB); mcs_nxt[k] = AH(SB); }
  half_t* mcur  = AH(SB);
  half_t* mnxt  = AH(SB);
  half_t* xprev = AH(SITES);
  size_t zend = cur;

  // non-zeroed scratch
  half_t* xcur  = AH(SITES);
  half_t* dxbuf = AH(SB);
  half_t* cmims = AH(SB);
  half_t* igbuf = AH(SB);
  half_t* obuf  = AH(SB);
  half_t* cat   = AH((size_t)SITES * 128);
  float* cc0 = AF((size_t)SITES * 256);
  float* cc1 = AF((size_t)SITES * 256);
  float* cc2 = AF((size_t)SITES * 256);
  float* part0 = AF(512); float* stat0 = AF(8);
  float* part1 = AF(512); float* stat1 = AF(8);
  float* part2 = AF(512); float* stat2 = AF(8);
  float* lpart = AF(256);
  float* parts[3] = {part0, part1, part2};
  float* stats[3] = {stat0, stat1, stat2};

  // ---- pack weights (f32 OIHW -> f16 [tap][Cout][Cin]) ----
  auto pack = [&](const float* w, half_t* dst, int Co, int Ci, int k2) {
    int total = Co * Ci * k2;
    pack_tap<<<dim3((total + 255) / 256), 256, 0, stream>>>(w, dst, Co, Ci, k2);
  };
  pack(F(STL + 3),  wp_stl_h, 256, 64, 25);
  pack(F(STL + 11), wp_stl_m, 256, 64, 25);
  pack(F(STL + 1),  wp_stl_c, 64, 128, 1);
  for (int j = 0; j < 3; ++j) {
    int mb = mimB(j), db = diffB(j), msb = mb + 12;
    pack(F(mb + 23), wp_mim_x[j], 256, 64, 25);
    pack(F(mb + 3),  wp_mim_h[j], 192, 64, 25);
    pack(F(mb + 11), wp_mim_m[j], 256, 64, 25);
    pack(F(mb + 1),  wp_mim_c[j], 64, 128, 1);
    pack(F(msb + 2), wp_mims_h[j], 256, 64, 25);
    pack(F(msb + 9), wp_mims_x[j], 256, 64, 25);
    pack(F(db + 2),  wp_diff_h[j], 256, 64, 25);
    pack(F(db + 9),  wp_diff_x[j], 256, 64, 25);
  }
  // zero states
  {
    size_t nbytes = zend - zstart;
    zero_u32<<<dim3(2048), 256, 0, stream>>>(
        (unsigned*)((char*)d_ws + zstart), nbytes / 4);
  }

  auto conv5 = [&](const half_t* in, const half_t* wp, const float* bias,
                   float* out, int Cout) {
    conv_wmma<25, 64><<<dim3(BATCH * HWD, Cout / 64), 64, 0, stream>>>(
        in, wp, bias, out, Cout);
  };
  auto conv1 = [&](const half_t* in, const half_t* wp, const float* bias,
                   float* out) {
    conv_wmma<1, 128><<<dim3(BATCH * HWD, 1), 64, 0, stream>>>(
        in, wp, bias, out, 64);
  };
  auto lnstats = [&](const float* buf, int Cc, int slot) {
    ln_part<<<dim3(64, BATCH), 256, 0, stream>>>(buf, HW * Cc, parts[slot]);
    ln_final<<<dim3(BATCH), 64, 0, stream>>>(parts[slot], stats[slot], HW * Cc);
  };
  const dim3 gEW(4096), bEW(256);   // SITES*64 threads
  const dim3 gS(64), bS(256);       // SITES threads
  float* gen_out = (float*)d_out;

  for (int t = 0; t < TSTEPS; ++t) {
    blend_x<<<gS, bS, 0, stream>>>(images, ss, xprev, xcur, t);

    // ---------- stlstm (layer 0) ----------
    conv_cin1<<<dim3(4096), 256, 0, stream>>>(xcur, F(STL + 13), F(STL + 12),
                                              cc0);
    lnstats(cc0, 256, 0);                                  // x_cc
    conv5(hs_cur[0], wp_stl_h, F(STL + 2), cc1, 256);
    lnstats(cc1, 256, 1);                                  // t_cc
    conv5(mcur, wp_stl_m, F(STL + 10), cc2, 256);
    lnstats(cc2, 256, 2);                                  // s_cc
    stlstm_gates<<<gEW, bEW, 0, stream>>>(
        cc0, cc1, cc2, stat0, stat1, stat2,
        F(STL + 9), F(STL + 8), F(STL + 5), F(STL + 4), F(STL + 7), F(STL + 6),
        cs_cur[0], mcur, cs_nxt[0], mnxt, cat, obuf);
    std::swap(mcur, mnxt);
    conv1(cat, wp_stl_c, F(STL + 0), cc2);                 // cell
    apply_out<<<gEW, bEW, 0, stream>>>(obuf, cc2, hs_nxt[0]);

    const half_t* dheff[3] = {nullptr, nullptr, nullptr};
    for (int i = 1; i < 4; ++i) {
      const int j = i - 1;
      const int mb = mimB(j), db = diffB(j), msb = mb + 12;
      // diff input
      const half_t* dxin;
      if (i == 1) {
        sub_f16<<<gEW, bEW, 0, stream>>>(hs_nxt[0], hs_cur[0], dxbuf);
        dxin = dxbuf;
      } else {
        dxin = dheff[j - 1];
      }
      // diff MIMN (skipped at t==0: reference discards the update)
      const half_t* dhe;
      if (t > 0) {
        conv5(dhs_cur[j], wp_diff_h[j], F(db + 1), cc0, 256);
        lnstats(cc0, 256, 0);
        conv5(dxin, wp_diff_x[j], F(db + 8), cc1, 256);
        lnstats(cc1, 256, 1);
        mimn_gates<<<gEW, bEW, 0, stream>>>(
            cc0, cc1, stat0, stat1,
            F(db + 4), F(db + 3), F(db + 6), F(db + 5),
            dcs_cur[j], F(db + 0), F(db + 7), dcs_nxt[j], dhs_nxt[j]);
        dhe = dhs_nxt[j];
      } else {
        dhe = dhs_cur[j];
      }
      dheff[j] = dhe;
      // MIMBlock gates
      conv5(hs_cur[i], wp_mim_h[j], F(mb + 2), cc0, 192);
      lnstats(cc0, 192, 0);                                // t_cc (3 gates)
      conv5(mcur, wp_mim_m[j], F(mb + 10), cc1, 256);
      lnstats(cc1, 256, 1);                                // s_cc
      conv5(hs_nxt[i - 1], wp_mim_x[j], F(mb + 22), cc2, 256);
      lnstats(cc2, 256, 2);                                // x_cc
      mimblock_gates<<<gEW, bEW, 0, stream>>>(
          cc0, cc1, cc2, stat0, stat1, stat2,
          F(mb + 5), F(mb + 4), F(mb + 7), F(mb + 6), F(mb + 9), F(mb + 8),
          mcur, mnxt, cat, igbuf, obuf);
      std::swap(mcur, mnxt);
      // MIMS: mimn_step(p.mims, x=dhe, h=cs_cur[i], c=mcs_cur[j])
      conv5(cs_cur[i], wp_mims_h[j], F(msb + 1), cc0, 256);
      lnstats(cc0, 256, 0);
      conv5(dhe, wp_mims_x[j], F(msb + 8), cc1, 256);
      lnstats(cc1, 256, 1);
      mimn_gates<<<gEW, bEW, 0, stream>>>(
          cc0, cc1, stat0, stat1,
          F(msb + 4), F(msb + 3), F(msb + 6), F(msb + 5),
          mcs_cur[j], F(msb + 0), F(msb + 7), mcs_nxt[j], cmims);
      finish_mim<<<gEW, bEW, 0, stream>>>(cmims, igbuf, cs_nxt[i], cat);
      conv1(cat, wp_mim_c[j], F(mb + 0), cc2);             // cell
      apply_out<<<gEW, bEW, 0, stream>>>(obuf, cc2, hs_nxt[i]);
    }
    // x_gen = 1x1 conv NH->1
    xgen_dot<<<gS, bS, 0, stream>>>(hs_nxt[3], F(XGEN + 1), F(XGEN + 0),
                                    gen_out, xprev, t);
    // swap state banks
    for (int k = 0; k < 4; ++k) { std::swap(hs_cur[k], hs_nxt[k]);
                                  std::swap(cs_cur[k], cs_nxt[k]); }
    for (int k = 0; k < 3; ++k) { std::swap(dhs_cur[k], dhs_nxt[k]);
                                  std::swap(dcs_cur[k], dcs_nxt[k]);
                                  std::swap(mcs_cur[k], mcs_nxt[k]); }
  }

  // loss = mean((gen - images[:,1:])^2), deterministic two-stage
  loss_part<<<dim3(256), 256, 0, stream>>>(gen_out, images, lpart);
  loss_final<<<dim3(1), 256, 0, stream>>>(lpart, gen_out + (out_size - 1));
}
